// GCNJK_79577154060352
// MI455X (gfx1250) — matrix operand order repaired and checked by reading, not compiled, not run on hardware
//
#include <hip/hip_runtime.h>

typedef __attribute__((ext_vector_type(16))) __bf16 v16bf;
typedef __attribute__((ext_vector_type(8)))  float  v8f;

#define EPS_BN 1e-5f

// ---------------------------------------------------------------------------
// WMMA GEMM: Out[nrows x NT*16] = A[nrows x 128] (x) W[128 x NT*16]
// A optionally fused as max(A0,A1) (JumpingKnowledge), bias optional.
// Block = 256 thr = 8 waves; each wave: 16 rows x all NT 16-col tiles.
// W staged in LDS pre-swizzled to the per-lane B-fragment layout
// (lane n<16 -> col n, K=0..7,16..23; lane n+16 -> col n, K=8..15,24..31,
//  matching the A fragment order) so a fragment = one 32B LDS vector load.
// Native __bf16 casts (RNE) so codegen can use packed bf16 converts.
// ---------------------------------------------------------------------------
template <int NT, bool JK, bool BIAS>
__global__ void __launch_bounds__(256)
gemm_wmma(const float* __restrict__ A0, const float* __restrict__ A1,
          const float* __restrict__ W, const float* __restrict__ bias,
          float* __restrict__ Out, int nrows)
{
  __shared__ __align__(32) __bf16 ldsW[4 * NT * 32 * 16];

  const int tid = threadIdx.x;
  const int ldw = NT * 16;

  // ---- stage W (fp32 global -> bf16 swizzled LDS) ----
  for (int idx = tid; idx < 4 * NT * 32; idx += 256) {
    int t   = idx / (NT * 32);
    int rem = idx - t * (NT * 32);
    int n   = rem >> 5;
    int l   = rem & 31;
    int col = n * 16 + (l & 15);
    int k0  = t * 32 + ((l >> 4) << 3);
    __bf16* dp = &ldsW[idx * 16];
#pragma unroll
    for (int j = 0; j < 16; ++j)
      dp[j] = (__bf16)W[(k0 + (j & 7) + ((j >> 3) << 4)) * ldw + col];
  }
  __syncthreads();

  const int wave  = tid >> 5;
  const int lane  = tid & 31;
  const int tileM = blockIdx.x * 128 + wave * 16;
  if (tileM >= nrows) return;              // wave-uniform; EXEC stays full

  const int m  = tileM + (lane & 15);
  const int hi = lane >> 4;

  v8f zero = {};
  v8f acc[NT];
#pragma unroll
  for (int n = 0; n < NT; ++n) acc[n] = zero;

  const float* rowA = A0 + (long)m * 128;
  const float* rowB = JK ? (A1 + (long)m * 128) : nullptr;

#pragma unroll
  for (int t = 0; t < 4; ++t) {
    // A fragment (ISA 16-bit A 16x32 layout): per lane two contiguous
    // 8-K runs at k = t*32 + hi*8 and k + 16.
    const int kA = t * 32 + hi * 8;
    float4 f0 = *(const float4*)(rowA + kA);
    float4 f1 = *(const float4*)(rowA + kA + 4);
    float4 f2 = *(const float4*)(rowA + kA + 16);
    float4 f3 = *(const float4*)(rowA + kA + 20);
    if (JK) {
      float4 g0 = *(const float4*)(rowB + kA);
      float4 g1 = *(const float4*)(rowB + kA + 4);
      float4 g2 = *(const float4*)(rowB + kA + 16);
      float4 g3 = *(const float4*)(rowB + kA + 20);
      f0.x = fmaxf(f0.x, g0.x); f0.y = fmaxf(f0.y, g0.y);
      f0.z = fmaxf(f0.z, g0.z); f0.w = fmaxf(f0.w, g0.w);
      f1.x = fmaxf(f1.x, g1.x); f1.y = fmaxf(f1.y, g1.y);
      f1.z = fmaxf(f1.z, g1.z); f1.w = fmaxf(f1.w, g1.w);
      f2.x = fmaxf(f2.x, g2.x); f2.y = fmaxf(f2.y, g2.y);
      f2.z = fmaxf(f2.z, g2.z); f2.w = fmaxf(f2.w, g2.w);
      f3.x = fmaxf(f3.x, g3.x); f3.y = fmaxf(f3.y, g3.y);
      f3.z = fmaxf(f3.z, g3.z); f3.w = fmaxf(f3.w, g3.w);
    }
    v16bf a;
    a[0]  = (__bf16)f0.x; a[1]  = (__bf16)f0.y;
    a[2]  = (__bf16)f0.z; a[3]  = (__bf16)f0.w;
    a[4]  = (__bf16)f1.x; a[5]  = (__bf16)f1.y;
    a[6]  = (__bf16)f1.z; a[7]  = (__bf16)f1.w;
    a[8]  = (__bf16)f2.x; a[9]  = (__bf16)f2.y;
    a[10] = (__bf16)f2.z; a[11] = (__bf16)f2.w;
    a[12] = (__bf16)f3.x; a[13] = (__bf16)f3.y;
    a[14] = (__bf16)f3.z; a[15] = (__bf16)f3.w;

#pragma unroll
    for (int n = 0; n < NT; ++n) {
      v16bf b = *(const v16bf*)&ldsW[((t * NT + n) * 32 + lane) * 16];
      acc[n] = __builtin_amdgcn_wmma_f32_16x16x32_bf16(
          false, a, false, b, (short)0, acc[n], false, false);
    }
  }

  // ---- epilogue: C/D layout (lane<16: col=lane, M=r; lane>=16: M=8+r) ----
  const int colBase = lane & 15;
  const int rOff    = (lane < 16) ? 0 : 8;
#pragma unroll
  for (int n = 0; n < NT; ++n) {
    int col  = n * 16 + colBase;
    float bv = BIAS ? bias[col] : 0.0f;
#pragma unroll
    for (int r = 0; r < 8; ++r) {
      int row = tileM + rOff + r;
      Out[(long)row * ldw + col] = acc[n][r] + bv;
    }
  }
}

// ---------------------------------------------------------------------------
// Graph / elementwise kernels
// ---------------------------------------------------------------------------
__global__ void init_deg_bn(float* deg, float* bn, int N) {
  int i = blockIdx.x * 256 + threadIdx.x;
  if (i < N) deg[i] = 1.0f;                       // self-loop weight
  if (blockIdx.x == 0) bn[threadIdx.x] = 0.0f;    // zero sum/sumsq (256)
}

__global__ void deg_edge(float* deg, const int* dst, const float* ew, int E) {
  int e = blockIdx.x * 256 + threadIdx.x;
  if (e < E) atomicAdd(&deg[dst[e]], ew[e]);
}

__global__ void deg_to_dinv(float* deg, int N) {
  int i = blockIdx.x * 256 + threadIdx.x;
  if (i < N) deg[i] = rsqrtf(deg[i]);
}

__global__ void edge_norm(float* nrm, const float* dinv, const int* src,
                          const int* dst, const float* ew, int E) {
  int e = blockIdx.x * 256 + threadIdx.x;
  if (e < E) nrm[e] = dinv[src[e]] * ew[e] * dinv[dst[e]];
}

// agg = bias + xw * dinv^2   (self-loop + bias init, before edge scatter)
__global__ void self_loop_init(float* agg, const float* xw, const float* bias,
                               const float* dinv, int NC) {
  int i = blockIdx.x * 256 + threadIdx.x;
  if (i < NC) {
    int n = i >> 7, c = i & 127;
    float di = dinv[n];
    agg[i] = bias[c] + xw[i] * di * di;
  }
}

// agg[dst] += xw[src] * norm  — one thread per (edge, channel), coalesced rows
__global__ void edge_scatter(float* agg, const float* __restrict__ xw,
                             const float* __restrict__ nrm,
                             const int* __restrict__ src,
                             const int* __restrict__ dst, int EC) {
  int i = blockIdx.x * 256 + threadIdx.x;
  if (i < EC) {
    int e = i >> 7, c = i & 127;
    float v = xw[((long)src[e] << 7) + c] * nrm[e];
    atomicAdd(&agg[((long)dst[e] << 7) + c], v);
  }
}

__global__ void bn_stats(const float* __restrict__ h, float* bn, int N) {
  int c  = threadIdx.x;                 // 128 threads, one channel each
  int r0 = blockIdx.x * 256;
  int r1 = r0 + 256 < N ? r0 + 256 : N;
  float s = 0.f, q = 0.f;
  for (int r = r0; r < r1; ++r) {
    float v = h[((long)r << 7) + c];
    s += v; q += v * v;
  }
  atomicAdd(&bn[c], s);
  atomicAdd(&bn[128 + c], q);
}

__global__ void bn_finalize(float* bn, const float* gamma, const float* beta,
                            int N) {
  int c = threadIdx.x;
  float inv_n = 1.0f / (float)N;
  float mu  = bn[c] * inv_n;
  float var = bn[128 + c] * inv_n - mu * mu;
  float a   = gamma[c] * rsqrtf(var + EPS_BN);
  bn[256 + c] = a;                 // scale
  bn[384 + c] = beta[c] - mu * a;  // shift
}

__global__ void bn_relu_apply(float* h, const float* __restrict__ bn, int NC) {
  int i = blockIdx.x * 256 + threadIdx.x;
  if (i < NC) {
    int c = i & 127;
    h[i] = fmaxf(h[i] * bn[256 + c] + bn[384 + c], 0.0f);
  }
}

// ---------------------------------------------------------------------------
extern "C" void kernel_launch(void* const* d_in, const int* in_sizes, int n_in,
                              void* d_out, int out_size, void* d_ws,
                              size_t ws_size, hipStream_t stream) {
  const float* x     = (const float*)d_in[0];
  const int*   ei    = (const int*)d_in[1];
  const float* ew    = (const float*)d_in[2];
  const float* W1    = (const float*)d_in[3];
  const float* b1    = (const float*)d_in[4];
  const float* gamma = (const float*)d_in[5];
  const float* beta  = (const float*)d_in[6];
  const float* W2    = (const float*)d_in[7];
  const float* b2    = (const float*)d_in[8];
  const float* Wf    = (const float*)d_in[9];
  const float* bf    = (const float*)d_in[10];
  float* out = (float*)d_out;

  const int N = in_sizes[0] / 128;
  const int E = in_sizes[2];
  const int* src = ei;        // edge_index[0]
  const int* dst = ei + E;    // edge_index[1]

  // workspace layout (floats): xw | h1 | agg2 | dinv | norm | bn(512)
  float* ws    = (float*)d_ws;
  const long NC = (long)N * 128;
  float* xw   = ws;
  float* h1   = ws + NC;        // agg1, then BN+ReLU in place
  float* agg2 = ws + 2 * NC;
  float* dinv = ws + 3 * NC;    // deg -> dinv in place
  float* nrm  = dinv + N;
  float* bn   = nrm + E;        // sum|sumsq|scale|shift

  const int EC = E * 128;
  dim3 blk(256);
  dim3 gN((N + 255) / 256), gE((E + 255) / 256);
  dim3 gNC(((int)NC + 255) / 256), gEC((EC + 255) / 256);
  dim3 gGemm((N + 127) / 128);

  // degree / normalization
  init_deg_bn<<<gN, blk, 0, stream>>>(dinv, bn, N);
  deg_edge<<<gE, blk, 0, stream>>>(dinv, dst, ew, E);
  deg_to_dinv<<<gN, blk, 0, stream>>>(dinv, N);
  edge_norm<<<gE, blk, 0, stream>>>(nrm, dinv, src, dst, ew, E);

  // layer 1: GCNConv -> BN -> ReLU
  gemm_wmma<8, false, false><<<gGemm, blk, 0, stream>>>(x, nullptr, W1, nullptr,
                                                        xw, N);
  self_loop_init<<<gNC, blk, 0, stream>>>(h1, xw, b1, dinv, (int)NC);
  edge_scatter<<<gEC, blk, 0, stream>>>(h1, xw, nrm, src, dst, EC);
  bn_stats<<<gN, dim3(128), 0, stream>>>(h1, bn, N);
  bn_finalize<<<dim3(1), dim3(128), 0, stream>>>(bn, gamma, beta, N);
  bn_relu_apply<<<gNC, blk, 0, stream>>>(h1, bn, (int)NC);

  // layer 2: GCNConv (no BN/ReLU)
  gemm_wmma<8, false, false><<<gGemm, blk, 0, stream>>>(h1, nullptr, W2,
                                                        nullptr, xw, N);
  self_loop_init<<<gNC, blk, 0, stream>>>(agg2, xw, b2, dinv, (int)NC);
  edge_scatter<<<gEC, blk, 0, stream>>>(agg2, xw, nrm, src, dst, EC);

  // JK max + final projection (fused into GEMM A-load)
  gemm_wmma<4, true, true><<<gGemm, blk, 0, stream>>>(h1, agg2, Wf, bf, out, N);
}
